// ALBERTAttention_56530359550281
// MI455X (gfx1250) — matrix operand order repaired and checked by reading, not compiled, not run on hardware
//
#include <hip/hip_runtime.h>
#include <stdint.h>

// ---------------------------------------------------------------------------
// CDNA5 (gfx1250) wave32 WMMA types
// ---------------------------------------------------------------------------
typedef __attribute__((ext_vector_type(16))) __bf16 v16bf;
typedef __attribute__((ext_vector_type(8)))  float  v8f;

union BF16Frag {
    v16bf v;
    uint4 q[2];   // 32 bytes = 16 bf16
};

__device__ __forceinline__ uint16_t f32_to_bf16(float f) {
    union { float f; uint32_t u; } x; x.f = f;
    uint32_t r = x.u + 0x7FFFu + ((x.u >> 16) & 1u);   // round-nearest-even
    return (uint16_t)(r >> 16);
}

__device__ __forceinline__ v8f v8f_zero() {
    v8f z;
#pragma unroll
    for (int i = 0; i < 8; ++i) z[i] = 0.0f;
    return z;
}

// A operand: 8 contiguous bf16 at +0 and 8 more at +16 elements
__device__ __forceinline__ BF16Frag load_a(const uint16_t* p) {
    BF16Frag f;
    f.q[0] = *(const uint4*)(p);
    f.q[1] = *(const uint4*)(p + 16);
    return f;
}
// B operand: 8 contiguous bf16 at +0 and 8 more at +16 elements
__device__ __forceinline__ BF16Frag load_b(const uint16_t* p) {
    BF16Frag f;
    f.q[0] = *(const uint4*)(p);
    f.q[1] = *(const uint4*)(p + 16);
    return f;
}

#define WMMA_BF16(A, B, C) \
    __builtin_amdgcn_wmma_f32_16x16x32_bf16(false, (A).v, false, (B).v, (short)0, (C), false, false)

// Problem constants
#define BATCH 2
#define SEQ   2048
#define DMODEL 1024
#define NHEAD 16
#define HDIM  64
#define MROWS (BATCH * SEQ)          // 4096
#define NCOLS (NHEAD * HDIM)         // 1024

// ---------------------------------------------------------------------------
// Prep kernels
// ---------------------------------------------------------------------------
__global__ void cast_bf16_kernel(const float* __restrict__ src,
                                 uint16_t* __restrict__ dst, int n) {
    int i = blockIdx.x * blockDim.x + threadIdx.x;
    if (i < n) dst[i] = f32_to_bf16(src[i]);
}

// W [D][N] fp32 -> WT [N][D] bf16 (coalesced read along N)
__global__ void transpose_cast_kernel(const float* __restrict__ W,
                                      uint16_t* __restrict__ WT) {
    int i = blockIdx.x * blockDim.x + threadIdx.x;   // [0, 1024*1024)
    int n = i & (NCOLS - 1);
    int d = i >> 10;
    WT[(size_t)n * DMODEL + d] = f32_to_bf16(W[i]);
}

__global__ void maskadd_kernel(const int* __restrict__ mask,
                               float* __restrict__ madd, int n) {
    int i = blockIdx.x * blockDim.x + threadIdx.x;
    if (i < n) madd[i] = (1.0f - (float)mask[i]) * -10000.0f;
}

// ---------------------------------------------------------------------------
// QKV projection GEMM: [4096 x 1024] x [1024 x 1024] (WT is N-major)
// 4 waves/block; each wave computes 32 rows x 64 cols (two A tiles reuse
// each B fragment).  K-loop step 32, software-pipelined with explicit
// ping-pong fragment buffers (manual 2x unroll -> no register rotation).
// mode 0: store bf16 [b,h,s,64]   (Q, K)
// mode 1: store bf16 [b,h,64,s]   (V transposed; packed b128 stores)
// ---------------------------------------------------------------------------
__global__ __launch_bounds__(128) void qkv_gemm_kernel(
    const uint16_t* __restrict__ X,     // [4096][1024] bf16 row-major
    const uint16_t* __restrict__ WT,    // [1024][1024] bf16 N-major
    const float* __restrict__ bias,     // [1024]
    uint16_t* __restrict__ out,
    int mode)
{
    const int lane = threadIdx.x & 31;
    const int wave = threadIdx.x >> 5;
    const int lo   = lane & 15;
    const int hi   = lane >> 4;
    const int m0   = blockIdx.x * 128 + wave * 32;   // rows m0..m0+31
    const int n0   = blockIdx.y * 64;

    v8f acc[2][4];
#pragma unroll
    for (int mi = 0; mi < 2; ++mi)
#pragma unroll
        for (int j = 0; j < 4; ++j) acc[mi][j] = v8f_zero();

    const uint16_t* a0p = X + (size_t)(m0 + lo) * DMODEL + hi * 8;
    const uint16_t* a1p = a0p + (size_t)16 * DMODEL;
    const uint16_t* bp[4];
#pragma unroll
    for (int j = 0; j < 4; ++j)
        bp[j] = WT + (size_t)(n0 + j * 16 + lo) * DMODEL + hi * 8;

    // ---- software pipeline: ping (phase A) / pong (phase B) buffers ----
    BF16Frag aA0, aA1, bA[4], aB0, aB1, bB[4];
    aA0 = load_a(a0p);
    aA1 = load_a(a1p);
#pragma unroll
    for (int j = 0; j < 4; ++j) bA[j] = load_b(bp[j]);

    for (int d0 = 0; d0 < DMODEL; d0 += 64) {
        // issue loads for d0+32 (phase B) before computing phase A
        aB0 = load_a(a0p + d0 + 32);
        aB1 = load_a(a1p + d0 + 32);
#pragma unroll
        for (int j = 0; j < 4; ++j) bB[j] = load_b(bp[j] + d0 + 32);

#pragma unroll
        for (int j = 0; j < 4; ++j) {
            acc[0][j] = WMMA_BF16(aA0, bA[j], acc[0][j]);
            acc[1][j] = WMMA_BF16(aA1, bA[j], acc[1][j]);
        }

        // issue loads for d0+64 (phase A of next iteration)
        if (d0 + 64 < DMODEL) {
            aA0 = load_a(a0p + d0 + 64);
            aA1 = load_a(a1p + d0 + 64);
#pragma unroll
            for (int j = 0; j < 4; ++j) bA[j] = load_b(bp[j] + d0 + 64);
        }

#pragma unroll
        for (int j = 0; j < 4; ++j) {
            acc[0][j] = WMMA_BF16(aB0, bB[j], acc[0][j]);
            acc[1][j] = WMMA_BF16(aB1, bB[j], acc[1][j]);
        }
    }

    // ---- epilogue: bias add, cast, store ----
#pragma unroll
    for (int mi = 0; mi < 2; ++mi) {
        const int mbase = m0 + mi * 16 + hi * 8;     // C layout: M = hi*8 + r
        const int b = mbase >> 11;
        const int s = mbase & (SEQ - 1);
#pragma unroll
        for (int j = 0; j < 4; ++j) {
            const int   n    = n0 + j * 16 + lo;
            const float bv   = bias[n];
            const int   hcol = n >> 6;
            const int   kcol = n & 63;
            if (mode == 0) {
                uint16_t* op =
                    out + ((size_t)(b * NHEAD + hcol) * SEQ + s) * HDIM + kcol;
#pragma unroll
                for (int r = 0; r < 8; ++r)
                    op[(size_t)r * HDIM] = f32_to_bf16(acc[mi][j][r] + bv);
            } else {
                uint16_t* op =
                    out + ((size_t)(b * NHEAD + hcol) * HDIM + kcol) * SEQ + s;
                uint32_t pk[4];
#pragma unroll
                for (int rr = 0; rr < 4; ++rr) {
                    uint32_t l0 = f32_to_bf16(acc[mi][j][2 * rr] + bv);
                    uint32_t h0 = f32_to_bf16(acc[mi][j][2 * rr + 1] + bv);
                    pk[rr] = l0 | (h0 << 16);
                }
                uint4 u; u.x = pk[0]; u.y = pk[1]; u.z = pk[2]; u.w = pk[3];
                *(uint4*)op = u;   // 8 consecutive t positions, 16B aligned
            }
        }
    }
}

// ---------------------------------------------------------------------------
// Flash attention: grid (F-tiles, B*H), 4 waves/block, each wave 16 F rows.
// T streamed in tiles of 32.  K fragments double-buffered across tiles; V
// fragments issued at the top of each tile so the softmax VALU chain hides
// their latency before the PV WMMAs.
// ---------------------------------------------------------------------------
__global__ __launch_bounds__(128) void flash_attn_kernel(
    const uint16_t* __restrict__ Q,       // [BH][2048][64] bf16
    const uint16_t* __restrict__ K,       // [BH][2048][64] bf16
    const uint16_t* __restrict__ Vt,      // [BH][64][2048] bf16
    const float*    __restrict__ maskadd, // [B*2048]
    float* __restrict__ out)              // [B][2048][16][64] fp32
{
    // per-wave P staging (C-layout f32 -> A-layout bf16), padded row stride 40
    __shared__ alignas(16) uint16_t stage[4][16 * 40];

    const int lane = threadIdx.x & 31;
    const int wave = threadIdx.x >> 5;
    const int lo   = lane & 15;
    const int hi   = lane >> 4;
    const int bh   = blockIdx.y;
    const int b    = bh >> 4;
    const int h    = bh & 15;
    const int f0   = blockIdx.x * 64 + wave * 16;

    // Q A-fragments for kdim 0..31 and 32..63 (register-resident all kernel)
    BF16Frag qa0, qa1;
    {
        const uint16_t* qrow = Q + ((size_t)bh * SEQ + f0 + lo) * HDIM + hi * 8;
        qa0 = load_a(qrow);
        qa1 = load_a(qrow + 32);
    }

    float mst[8], lst[8];
#pragma unroll
    for (int r = 0; r < 8; ++r) { mst[r] = -1e30f; lst[r] = 0.0f; }
    v8f o[4];
#pragma unroll
    for (int j = 0; j < 4; ++j) o[j] = v8f_zero();

    const uint16_t* Kbase = K  + (size_t)bh * SEQ * HDIM;
    const uint16_t* Vbase = Vt + (size_t)bh * HDIM * SEQ;
    const float*    mbase = maskadd + b * SEQ;
    uint16_t*       st    = stage[wave];

    // one 32-wide T tile; kc = current K fragments, kn = loaded for t0+32
    auto process_tile = [&](int t0, BF16Frag (&kc)[4], BF16Frag (&kn)[4],
                            bool load_next) {
        // ---- issue V fragment loads for THIS tile early ----
        BF16Frag vf[4];
#pragma unroll
        for (int j = 0; j < 4; ++j)
            vf[j] = load_b(Vbase + (size_t)(j * 16 + lo) * SEQ + t0 + hi * 8);

        // ---- issue K fragment loads for NEXT tile ----
        if (load_next) {
#pragma unroll
            for (int jt = 0; jt < 2; ++jt) {
                const uint16_t* krow =
                    Kbase + (size_t)(t0 + 32 + jt * 16 + lo) * HDIM + hi * 8;
                kn[jt * 2 + 0] = load_b(krow);        // kdim  0..31
                kn[jt * 2 + 1] = load_b(krow + 32);   // kdim 32..63
            }
            // distance-2 prefetch (global_prefetch_b8) for K and V tiles
            if (t0 + 64 < SEQ) {
                __builtin_prefetch(Kbase + (size_t)(t0 + 64 + lane) * HDIM, 0, 0);
                __builtin_prefetch(Vbase + (size_t)lane * SEQ + t0 + 64, 0, 0);
            }
        }

        // ---- logits: S[16x32] = Q[16x64] * K^T ----
        v8f s[2];
#pragma unroll
        for (int jt = 0; jt < 2; ++jt) {
            v8f a = v8f_zero();
            a = WMMA_BF16(qa0, kc[jt * 2 + 0], a);
            a = WMMA_BF16(qa1, kc[jt * 2 + 1], a);
            s[jt] = a;
        }

        // ---- scale + key-padding mask (column uniform per lane) ----
        const float madd0 = mbase[t0 + lo];
        const float madd1 = mbase[t0 + 16 + lo];
        float rmax[8];
#pragma unroll
        for (int r = 0; r < 8; ++r) {
            s[0][r] = s[0][r] * 0.125f + madd0;    // 1/sqrt(64)
            s[1][r] = s[1][r] * 0.125f + madd1;
            rmax[r] = fmaxf(s[0][r], s[1][r]);
        }
#pragma unroll
        for (int x = 1; x < 16; x <<= 1) {
#pragma unroll
            for (int r = 0; r < 8; ++r)
                rmax[r] = fmaxf(rmax[r], __shfl_xor(rmax[r], x, 32));
        }

        // ---- online softmax update ----
        float alpha[8], rsum[8];
#pragma unroll
        for (int r = 0; r < 8; ++r) {
            const float mnew = fmaxf(mst[r], rmax[r]);
            alpha[r] = __expf(mst[r] - mnew);
            mst[r]   = mnew;
            s[0][r]  = __expf(s[0][r] - mnew);
            s[1][r]  = __expf(s[1][r] - mnew);
            rsum[r]  = s[0][r] + s[1][r];
        }
#pragma unroll
        for (int x = 1; x < 16; x <<= 1) {
#pragma unroll
            for (int r = 0; r < 8; ++r)
                rsum[r] += __shfl_xor(rsum[r], x, 32);
        }
#pragma unroll
        for (int r = 0; r < 8; ++r) lst[r] = lst[r] * alpha[r] + rsum[r];
#pragma unroll
        for (int j = 0; j < 4; ++j)
#pragma unroll
            for (int r = 0; r < 8; ++r) o[j][r] *= alpha[r];

        // ---- re-layout P: C-layout f32 -> A-layout bf16 via LDS ----
#pragma unroll
        for (int jt = 0; jt < 2; ++jt)
#pragma unroll
            for (int r = 0; r < 8; ++r)
                st[(hi * 8 + r) * 40 + jt * 16 + lo] = f32_to_bf16(s[jt][r]);

        asm volatile("s_wait_dscnt 0" ::: "memory");   // DS RAW fence (same wave)

        BF16Frag pa;   // A operand: row = lo, t chunks at hi*8 and 16+hi*8
        pa.q[0] = *(const uint4*)(&st[lo * 40 + hi * 8]);
        pa.q[1] = *(const uint4*)(&st[lo * 40 + 16 + hi * 8]);

        // ---- O[16x64] += P[16x32] * V[32x64] ----
#pragma unroll
        for (int j = 0; j < 4; ++j)
            o[j] = WMMA_BF16(pa, vf[j], o[j]);
    };

    // preload K fragments for t0 = 0
    BF16Frag kbufA[4], kbufB[4];
#pragma unroll
    for (int jt = 0; jt < 2; ++jt) {
        const uint16_t* krow = Kbase + (size_t)(jt * 16 + lo) * HDIM + hi * 8;
        kbufA[jt * 2 + 0] = load_b(krow);
        kbufA[jt * 2 + 1] = load_b(krow + 32);
    }

    for (int t0 = 0; t0 < SEQ; t0 += 64) {
        process_tile(t0,      kbufA, kbufB, true);
        process_tile(t0 + 32, kbufB, kbufA, t0 + 64 < SEQ);
    }

    // ---- normalize and store out[b][f][h][k] ----
#pragma unroll
    for (int j = 0; j < 4; ++j) {
#pragma unroll
        for (int r = 0; r < 8; ++r) {
            const int f = f0 + hi * 8 + r;
            const int k = j * 16 + lo;
            out[(((size_t)b * SEQ + f) * NHEAD + h) * HDIM + k] = o[j][r] / lst[r];
        }
    }
}

// ---------------------------------------------------------------------------
// Launcher
// ---------------------------------------------------------------------------
extern "C" void kernel_launch(void* const* d_in, const int* in_sizes, int n_in,
                              void* d_out, int out_size, void* d_ws, size_t ws_size,
                              hipStream_t stream) {
    (void)in_sizes; (void)n_in; (void)out_size; (void)ws_size;

    const float* from_t = (const float*)d_in[0];
    const float* to_t   = (const float*)d_in[1];
    const int*   amask  = (const int*)d_in[2];
    const float* Wq     = (const float*)d_in[3];
    const float* bq     = (const float*)d_in[4];
    const float* Wk     = (const float*)d_in[5];
    const float* bk     = (const float*)d_in[6];
    const float* Wv     = (const float*)d_in[7];
    const float* bv     = (const float*)d_in[8];
    float* out = (float*)d_out;

    char* ws = (char*)d_ws;
    uint16_t* Xf   = (uint16_t*)(ws);                          //  8 MiB
    uint16_t* Xt   = (uint16_t*)(ws + (size_t)( 8u << 20));    //  8 MiB
    uint16_t* WqT  = (uint16_t*)(ws + (size_t)(16u << 20));    //  2 MiB
    uint16_t* WkT  = (uint16_t*)(ws + (size_t)(18u << 20));    //  2 MiB
    uint16_t* WvT  = (uint16_t*)(ws + (size_t)(20u << 20));    //  2 MiB
    float*    madd = (float*)   (ws + (size_t)(22u << 20));    // 16 KiB
    uint16_t* Qw   = (uint16_t*)(ws + (size_t)(23u << 20));    //  8 MiB
    uint16_t* Kw   = (uint16_t*)(ws + (size_t)(31u << 20));    //  8 MiB
    uint16_t* Vtw  = (uint16_t*)(ws + (size_t)(39u << 20));    //  8 MiB (47 MiB total)

    const int NX = MROWS * DMODEL;   // 4096*1024
    cast_bf16_kernel<<<NX / 256, 256, 0, stream>>>(from_t, Xf, NX);
    cast_bf16_kernel<<<NX / 256, 256, 0, stream>>>(to_t,   Xt, NX);

    const int NW = DMODEL * NCOLS;   // 1024*1024
    transpose_cast_kernel<<<NW / 256, 256, 0, stream>>>(Wq, WqT);
    transpose_cast_kernel<<<NW / 256, 256, 0, stream>>>(Wk, WkT);
    transpose_cast_kernel<<<NW / 256, 256, 0, stream>>>(Wv, WvT);

    maskadd_kernel<<<(BATCH * SEQ) / 256, 256, 0, stream>>>(amask, madd, BATCH * SEQ);

    dim3 ggrid(MROWS / 128, NCOLS / 64);
    qkv_gemm_kernel<<<ggrid, 128, 0, stream>>>(Xf, WqT, bq, Qw,  0);
    qkv_gemm_kernel<<<ggrid, 128, 0, stream>>>(Xt, WkT, bk, Kw,  0);
    qkv_gemm_kernel<<<ggrid, 128, 0, stream>>>(Xt, WvT, bv, Vtw, 1);

    dim3 fgrid(SEQ / 64, BATCH * NHEAD);
    flash_attn_kernel<<<fgrid, 128, 0, stream>>>(Qw, Kw, Vtw, madd, out);
}
